// GATwithMinCutPooling_23648089931944
// MI455X (gfx1250) — matrix operand order repaired and checked by reading, not compiled, not run on hardware
//
#include <hip/hip_runtime.h>
#include <climits>

typedef __attribute__((ext_vector_type(16))) _Float16 v16h;
typedef __attribute__((ext_vector_type(8)))  _Float16 v8h;
typedef __attribute__((ext_vector_type(8)))  float    v8f;

#define SLOPE 0.2f
#define EPSK  1e-15f
#define KPAD  10240   // 10000 padded to multiple of 32*20

// ---------- helpers ----------
__device__ __forceinline__ int f2ord(float f) {
    int i = __float_as_int(f);
    return (i >= 0) ? i : (i ^ 0x7FFFFFFF);
}
__device__ __forceinline__ float ord2f(int i) {
    return __int_as_float((i >= 0) ? i : (i ^ 0x7FFFFFFF));
}

// ---------- utility kernels ----------
__global__ void k_fill_i32(int* p, int v, int n) {
    int i = blockIdx.x * blockDim.x + threadIdx.x;
    if (i < n) p[i] = v;
}

__global__ void k_f32_to_f16(const float* __restrict__ s, _Float16* __restrict__ d, int n) {
    int i = blockIdx.x * blockDim.x + threadIdx.x;
    if (i < n) d[i] = (_Float16)s[i];
}

// Pack row-major f32 B[K,N] into per-lane WMMA B fragments (f16):
// Bp[(nt*KT + kt)*512 + lane*16 + e] = B[(kt*32 + (lane>>4)*16 + e)*N + nt*16 + (lane&15)]
__global__ void k_pack_b(const float* __restrict__ B, _Float16* __restrict__ Bp,
                         int Kdim, int Ndim) {
    int i = blockIdx.x * blockDim.x + threadIdx.x;
    if (i >= Kdim * Ndim) return;
    int KT = Kdim >> 5;
    int tIdx = i >> 9;            // 512 halfs per 32x16 tile
    int r    = i & 511;
    int lane = r >> 4, e = r & 15;
    int nt = tIdx / KT, kt = tIdx - nt * KT;
    int hs = lane >> 4, l15 = lane & 15;
    int srcK = kt * 32 + hs * 16 + e;
    int srcN = nt * 16 + l15;
    Bp[i] = (_Float16)B[(size_t)srcK * Ndim + srcN];
}

// Pack s^T [16, KPAD] f16 (zero-padded tail)
__global__ void k_pack_sT(const float* __restrict__ s, _Float16* __restrict__ sT, int n) {
    int i = blockIdx.x * blockDim.x + threadIdx.x;
    if (i >= 16 * KPAD) return;
    int m = i / KPAD, nn = i - m * KPAD;
    sT[i] = (nn < n) ? (_Float16)s[(size_t)nn * 16 + m] : (_Float16)0.f;
}

// Pack x^T [256, KPAD] f16 (zero-padded tail)
__global__ void k_pack_xT(const float* __restrict__ x, _Float16* __restrict__ xT, int n) {
    int i = blockIdx.x * blockDim.x + threadIdx.x;
    if (i >= 256 * KPAD) return;
    int f = i / KPAD, nn = i - f * KPAD;
    xT[i] = (nn < n) ? (_Float16)x[(size_t)nn * 256 + f] : (_Float16)0.f;
}

// ---------- WMMA GEMM: C[M,N] = A[M,K] * Bp, K = 32*KS (compile-time) ----------
// one wave per 16x16 output tile; A row-major f16, B pre-packed fragments
template<int KS>
__global__ void k_wmma_gemm(const _Float16* __restrict__ A,
                            const _Float16* __restrict__ Bp,
                            float* __restrict__ C, int M, int N) {
    const int K = KS * 32;
    int m0 = blockIdx.x * 16;
    int lane = threadIdx.x;
    int hs = lane >> 4, l15 = lane & 15;
    const _Float16* arow0 = A + (size_t)(m0 + l15) * K + hs * 16;
    const _Float16* fb = Bp + ((size_t)blockIdx.y * KS) * 512 + lane * 16;
    v8f acc = {};
    #pragma unroll
    for (int kt = 0; kt < KS; ++kt) {
        v8h a0 = *(const v8h*)(arow0 + kt * 32);
        v8h a1 = *(const v8h*)(arow0 + kt * 32 + 8);
        v16h a = __builtin_shufflevector(a0, a1, 0,1,2,3,4,5,6,7,8,9,10,11,12,13,14,15);
        v16h b = *(const v16h*)(fb + kt * 512);
        acc = __builtin_amdgcn_wmma_f32_16x16x32_f16(false, a, false, b,
                                                     (short)0, acc, false, false);
    }
    float* crow = C + (size_t)(m0 + hs * 8) * N + blockIdx.y * 16 + l15;
    #pragma unroll
    for (int r = 0; r < 8; ++r) crow[(size_t)r * N] = acc[r];
}

// ---------- out = s^T x via WMMA, split-K with atomic reduction ----------
// A = sT [16, KPAD], B^T = xT [256, KPAD]; grid (16 n-tiles, 16 k-splits)
#define STX_KSW 20
__global__ void k_wmma_stx(const _Float16* __restrict__ sT,
                           const _Float16* __restrict__ xT,
                           float* __restrict__ oacc) {
    int n0 = blockIdx.x * 16;
    int k0 = blockIdx.y * (STX_KSW * 32);
    int lane = threadIdx.x;
    int hs = lane >> 4, l15 = lane & 15;
    const _Float16* arow0 = sT + (size_t)l15 * KPAD + k0 + hs * 16;
    const _Float16* brow0 = xT + (size_t)(n0 + l15) * KPAD + k0 + hs * 16;
    v8f acc = {};
    #pragma unroll
    for (int kt = 0; kt < STX_KSW; ++kt) {
        v8h a0 = *(const v8h*)(arow0 + kt * 32);
        v8h a1 = *(const v8h*)(arow0 + kt * 32 + 8);
        v16h a = __builtin_shufflevector(a0, a1, 0,1,2,3,4,5,6,7,8,9,10,11,12,13,14,15);
        v16h b = *(const v16h*)(brow0 + kt * 32);
        acc = __builtin_amdgcn_wmma_f32_16x16x32_f16(false, a, false, b,
                                                     (short)0, acc, false, false);
    }
    #pragma unroll
    for (int r = 0; r < 8; ++r)
        atomicAdd(&oacc[(r + hs * 8) * 256 + n0 + l15], acc[r]);
}

// ---------- attention coefficients: a = sum_c h[n,h,c]*att[h,c] ----------
__global__ void k_att(const float* __restrict__ h,
                      const float* __restrict__ att_s, const float* __restrict__ att_d,
                      float* __restrict__ a_s, float* __restrict__ a_d,
                      int n, int H, int C) {
    int i = blockIdx.x * blockDim.x + threadIdx.x;
    if (i >= n * H) return;
    int node = i / H, hd = i - node * H;
    const float* hp = h + (size_t)node * H * C + hd * C;
    float s = 0.f, d = 0.f;
    for (int c = 0; c < C; ++c) {
        float v = hp[c];
        s += v * att_s[hd * C + c];
        d += v * att_d[hd * C + c];
    }
    a_s[i] = s; a_d[i] = d;
}

// ---------- edge passes (self-loops appended as e in [E, E+n)) ----------
__global__ void k_edge_max(const int* __restrict__ src, const int* __restrict__ dst,
                           int E, int n, int H,
                           const float* __restrict__ a_s, const float* __restrict__ a_d,
                           int* __restrict__ segmax) {
    int i = blockIdx.x * blockDim.x + threadIdx.x;
    int ET = E + n;
    if (i >= ET * H) return;
    int e = i / H, hd = i - e * H;
    int sN = (e < E) ? src[e] : (e - E);
    int dN = (e < E) ? dst[e] : (e - E);
    float ev = a_s[sN * H + hd] + a_d[dN * H + hd];
    ev = (ev > 0.f) ? ev : SLOPE * ev;
    atomicMax(&segmax[dN * H + hd], f2ord(ev));
}

__global__ void k_edge_sum(const int* __restrict__ src, const int* __restrict__ dst,
                           int E, int n, int H,
                           const float* __restrict__ a_s, const float* __restrict__ a_d,
                           const int* __restrict__ segmax, float* __restrict__ segsum) {
    int i = blockIdx.x * blockDim.x + threadIdx.x;
    int ET = E + n;
    if (i >= ET * H) return;
    int e = i / H, hd = i - e * H;
    int sN = (e < E) ? src[e] : (e - E);
    int dN = (e < E) ? dst[e] : (e - E);
    int di = dN * H + hd;
    float ev = a_s[sN * H + hd] + a_d[di];
    ev = (ev > 0.f) ? ev : SLOPE * ev;
    atomicAdd(&segsum[di], __expf(ev - ord2f(segmax[di])));
}

__global__ void k_edge_aggr(const int* __restrict__ src, const int* __restrict__ dst,
                            int E, int n, int H, int C,
                            const float* __restrict__ a_s, const float* __restrict__ a_d,
                            const int* __restrict__ segmax, const float* __restrict__ segsum,
                            const float* __restrict__ hfeat, float* __restrict__ out) {
    int i = blockIdx.x * blockDim.x + threadIdx.x;
    int ET = E + n;
    if (i >= ET * H) return;
    int e = i / H, hd = i - e * H;
    int sN = (e < E) ? src[e] : (e - E);
    int dN = (e < E) ? dst[e] : (e - E);
    int di = dN * H + hd;
    float ev = a_s[sN * H + hd] + a_d[di];
    ev = (ev > 0.f) ? ev : SLOPE * ev;
    float alpha = __expf(ev - ord2f(segmax[di])) / (segsum[di] + 1e-16f);
    const float* hp = hfeat + (size_t)sN * H * C + hd * C;
    float* op = out + (size_t)dN * H * C + hd * C;
    for (int c = 0; c < C; ++c)
        atomicAdd(&op[c], hp[c] * alpha);
}

// ---------- ELU(out1 + b1) -> f16 ----------
__global__ void k_elu_to_h16(const float* __restrict__ in, const float* __restrict__ b,
                             _Float16* __restrict__ out, int n, int F) {
    int i = blockIdx.x * blockDim.x + threadIdx.x;
    if (i >= n * F) return;
    float v = in[i] + b[i % F];
    v = (v > 0.f) ? v : (__expf(v) - 1.f);
    out[i] = (_Float16)v;
}

__global__ void k_add_bias(float* __restrict__ p, const float* __restrict__ b, int n, int F) {
    int i = blockIdx.x * blockDim.x + threadIdx.x;
    if (i < n) p[i] += b[i % F];
}

// ---------- s = softmax(softmax(xg @ Ws + bs)) ----------
__global__ void k_softmax_s(const float* __restrict__ xg, const float* __restrict__ Ws,
                            const float* __restrict__ bs, float* __restrict__ s, int n) {
    int i = blockIdx.x * blockDim.x + threadIdx.x;
    if (i >= n) return;
    float row[16], lg[16];
    #pragma unroll
    for (int k = 0; k < 16; ++k) row[k] = xg[(size_t)i * 16 + k];
    #pragma unroll
    for (int l = 0; l < 16; ++l) {
        float t = bs[l];
        #pragma unroll
        for (int k = 0; k < 16; ++k) t += row[k] * Ws[k * 16 + l];
        lg[l] = t;
    }
    #pragma unroll
    for (int pass = 0; pass < 2; ++pass) {
        float m = lg[0];
        #pragma unroll
        for (int l = 1; l < 16; ++l) m = fmaxf(m, lg[l]);
        float sum = 0.f;
        #pragma unroll
        for (int l = 0; l < 16; ++l) { lg[l] = __expf(lg[l] - m); sum += lg[l]; }
        float inv = 1.f / sum;
        #pragma unroll
        for (int l = 0; l < 16; ++l) lg[l] *= inv;
    }
    #pragma unroll
    for (int l = 0; l < 16; ++l) s[(size_t)i * 16 + l] = lg[l];
}

// ---------- out-degree (adj.sum(-1)) over original E edges ----------
__global__ void k_deg(const int* __restrict__ src, float* __restrict__ deg, int E) {
    int i = blockIdx.x * blockDim.x + threadIdx.x;
    if (i < E) atomicAdd(&deg[src[i]], 1.0f);
}

// ---------- out_adj_raw = s^T A s via per-edge rank-1, LDS staged ----------
#define OA_CHUNK 128
#define OA_EPB   2560
__global__ void k_outadj(const float* __restrict__ s,
                         const int* __restrict__ src, const int* __restrict__ dst,
                         int E, float* __restrict__ oadj) {
    __shared__ float ssm[OA_CHUNK * 16];
    __shared__ float sdm[OA_CHUNK * 16];
    int t = threadIdx.x, k = t >> 4, l = t & 15;
    float acc = 0.f;
    int base = blockIdx.x * OA_EPB;
    int end = base + OA_EPB; if (end > E) end = E;
    for (int c0 = base; c0 < end; c0 += OA_CHUNK) {
        int cnt = end - c0; if (cnt > OA_CHUNK) cnt = OA_CHUNK;
        __syncthreads();
        for (int i = t; i < cnt * 16; i += 256) {
            int e = i >> 4, kk = i & 15;
            ssm[i] = s[(size_t)src[c0 + e] * 16 + kk];
            sdm[i] = s[(size_t)dst[c0 + e] * 16 + kk];
        }
        __syncthreads();
        for (int e = 0; e < cnt; ++e)
            acc += ssm[e * 16 + k] * sdm[e * 16 + l];
    }
    atomicAdd(&oadj[t], acc);
}

// ---------- ss = s^T s and mincut_den = sum_n deg[n]*||s_n||^2 ----------
#define SS_NPB 2560
__global__ void k_ss_den(const float* __restrict__ s, const float* __restrict__ deg,
                         int n, float* __restrict__ ssb, float* __restrict__ den) {
    __shared__ float sn[OA_CHUNK * 16];
    __shared__ float dg[OA_CHUNK];
    int t = threadIdx.x, k = t >> 4, l = t & 15;
    float acc = 0.f, dacc = 0.f;
    int base = blockIdx.x * SS_NPB;
    int end = base + SS_NPB; if (end > n) end = n;
    for (int c0 = base; c0 < end; c0 += OA_CHUNK) {
        int cnt = end - c0; if (cnt > OA_CHUNK) cnt = OA_CHUNK;
        __syncthreads();
        for (int i = t; i < cnt * 16; i += 256)
            sn[i] = s[(size_t)(c0 + (i >> 4)) * 16 + (i & 15)];
        for (int i = t; i < cnt; i += 256)
            dg[i] = deg[c0 + i];
        __syncthreads();
        for (int e = 0; e < cnt; ++e) {
            float a = sn[e * 16 + k];
            acc += a * sn[e * 16 + l];
            if (k == l) dacc += dg[e] * a * a;
        }
    }
    atomicAdd(&ssb[t], acc);
    if (k == l) atomicAdd(den, dacc);
}

// ---------- finalize losses + normalized out_adj + copy out ----------
__global__ void k_finalize(const float* __restrict__ oadj, const float* __restrict__ ssb,
                           const float* __restrict__ den, const float* __restrict__ out_acc,
                           float* __restrict__ d_out) {
    const int OFF_OUT = 160000, OFF_ADJ = 164096, OFF_ML = 164352, OFF_OL = 164353;
    __shared__ float adj[256];
    __shared__ float red[256];
    __shared__ float ddv[16];
    __shared__ float snorm_sh;
    int t = threadIdx.x;
    for (int i = t; i < 4096; i += 256) d_out[OFF_OUT + i] = out_acc[i];

    adj[t] = oadj[t];
    float ssv = ssb[t];
    red[t] = ssv * ssv;
    __syncthreads();
    for (int sft = 128; sft > 0; sft >>= 1) {
        if (t < sft) red[t] += red[t + sft];
        __syncthreads();
    }
    if (t == 0) snorm_sh = sqrtf(red[0]);
    __syncthreads();
    float diff = ssv / snorm_sh - (((t >> 4) == (t & 15)) ? 0.25f : 0.f);
    red[t] = diff * diff;
    __syncthreads();
    for (int sft = 128; sft > 0; sft >>= 1) {
        if (t < sft) red[t] += red[t + sft];
        __syncthreads();
    }
    if (t == 0) d_out[OFF_OL] = sqrtf(red[0]);

    if (t == 0) {
        float num = 0.f;
        for (int kk = 0; kk < 16; ++kk) num += adj[kk * 17];
        d_out[OFF_ML] = -(num / den[0]);
    }
    __syncthreads();
    if ((t >> 4) == (t & 15)) adj[t] = 0.f;
    __syncthreads();
    if (t < 16) {
        float rs = 0.f;
        for (int l = 0; l < 16; ++l) rs += adj[t * 16 + l];
        ddv[t] = sqrtf(rs) + EPSK;
    }
    __syncthreads();
    d_out[OFF_ADJ + t] = adj[t] / (ddv[t & 15] * ddv[t >> 4]);
}

// ---------------------------------------------------------------------------
extern "C" void kernel_launch(void* const* d_in, const int* in_sizes, int n_in,
                              void* d_out_v, int out_size, void* d_ws, size_t ws_size,
                              hipStream_t stream) {
    const int N = 10000, E = 320000, FIN = 256, F1 = 64, H1 = 8, C1 = 8, K = 16;
    const int ET = E + N;

    const float* x        = (const float*)d_in[0];
    const int*   ei       = (const int*)d_in[1];
    const float* W1       = (const float*)d_in[2];
    const float* att_src1 = (const float*)d_in[3];
    const float* att_dst1 = (const float*)d_in[4];
    const float* b1       = (const float*)d_in[5];
    const float* W2       = (const float*)d_in[6];
    const float* att_src2 = (const float*)d_in[7];
    const float* att_dst2 = (const float*)d_in[8];
    const float* b2       = (const float*)d_in[9];
    const float* Ws       = (const float*)d_in[10];
    const float* bs       = (const float*)d_in[11];
    const int* src = ei;
    const int* dst = ei + E;
    float* d_out = (float*)d_out_v;

    char* ws = (char*)d_ws;
    size_t off = 0;
    auto alloc = [&](size_t bytes) -> char* {
        char* p = ws + off;
        off += (bytes + 255) & ~(size_t)255;
        return p;
    };
    _Float16* xh    = (_Float16*)alloc((size_t)N * FIN * 2);   // x f16 row-major (GEMM1 A)
    _Float16* xT    = (_Float16*)alloc((size_t)FIN * KPAD * 2);// x^T f16 padded (STX B)
    _Float16* sT    = (_Float16*)alloc((size_t)K * KPAD * 2);  // s^T f16 padded (STX A)
    _Float16* w1p   = (_Float16*)alloc((size_t)FIN * F1 * 2);  // packed B fragments
    _Float16* w2p   = (_Float16*)alloc((size_t)F1 * K * 2);
    float*    h1    = (float*)alloc((size_t)N * F1 * 4);
    float*    out1  = (float*)alloc((size_t)N * F1 * 4);
    _Float16* hacth = (_Float16*)alloc((size_t)N * F1 * 2);
    float*    as1   = (float*)alloc((size_t)N * H1 * 4);
    float*    ad1   = (float*)alloc((size_t)N * H1 * 4);
    int*      smax1 = (int*)alloc((size_t)N * H1 * 4);
    float*    ssum1 = (float*)alloc((size_t)N * H1 * 4);
    float*    xgl   = (float*)alloc((size_t)N * K * 4);
    float*    as2   = (float*)alloc((size_t)N * 4);
    float*    ad2   = (float*)alloc((size_t)N * 4);
    int*      smax2 = (int*)alloc((size_t)N * 4);
    float*    ssum2 = (float*)alloc((size_t)N * 4);
    float*    sbuf  = (float*)alloc((size_t)N * K * 4);
    float*    deg   = (float*)alloc((size_t)N * 4);
    float*    oacc  = (float*)alloc((size_t)K * FIN * 4);
    float*    oadj  = (float*)alloc(256 * 4);
    float*    ssb   = (float*)alloc(256 * 4);
    float*    denb  = (float*)alloc(256);

    // ---- init accumulators (inside graph: deterministic on replay) ----
    hipMemsetAsync(ssum1, 0, (size_t)N * H1 * 4, stream);
    hipMemsetAsync(out1,  0, (size_t)N * F1 * 4, stream);
    hipMemsetAsync(ssum2, 0, (size_t)N * 4, stream);
    hipMemsetAsync(d_out, 0, (size_t)N * K * 4, stream);   // xg accumulation region
    hipMemsetAsync(deg,   0, (size_t)N * 4, stream);
    hipMemsetAsync(oacc,  0, (size_t)K * FIN * 4, stream);
    hipMemsetAsync(oadj,  0, 256 * 4, stream);
    hipMemsetAsync(ssb,   0, 256 * 4, stream);
    hipMemsetAsync(denb,  0, 4, stream);
    k_fill_i32<<<(N * H1 + 255) / 256, 256, 0, stream>>>(smax1, INT_MIN, N * H1);
    k_fill_i32<<<(N + 255) / 256, 256, 0, stream>>>(smax2, INT_MIN, N);

    // ---- f16 conversions / fragment packing ----
    k_f32_to_f16<<<((size_t)N * FIN + 255) / 256, 256, 0, stream>>>(x, xh, N * FIN);
    k_pack_xT<<<((size_t)FIN * KPAD + 255) / 256, 256, 0, stream>>>(x, xT, N);
    k_pack_b<<<(FIN * F1 + 255) / 256, 256, 0, stream>>>(W1, w1p, FIN, F1);
    k_pack_b<<<(F1 * K + 255) / 256, 256, 0, stream>>>(W2, w2p, F1, K);

    // ---- layer 1: h1 = x @ W1 (WMMA, K=256 -> 8 unrolled wmma) ----
    k_wmma_gemm<8><<<dim3(N / 16, F1 / 16), 32, 0, stream>>>(xh, w1p, h1, N, F1);
    k_att<<<(N * H1 + 255) / 256, 256, 0, stream>>>(h1, att_src1, att_dst1, as1, ad1, N, H1, C1);
    {
        int nt = ET * H1, gb = (nt + 255) / 256;
        k_edge_max <<<gb, 256, 0, stream>>>(src, dst, E, N, H1, as1, ad1, smax1);
        k_edge_sum <<<gb, 256, 0, stream>>>(src, dst, E, N, H1, as1, ad1, smax1, ssum1);
        k_edge_aggr<<<gb, 256, 0, stream>>>(src, dst, E, N, H1, C1, as1, ad1, smax1, ssum1, h1, out1);
    }
    k_elu_to_h16<<<((size_t)N * F1 + 255) / 256, 256, 0, stream>>>(out1, b1, hacth, N, F1);

    // ---- layer 2: xgl = elu(h) @ W2 (WMMA, K=64 -> 2 unrolled wmma) ----
    k_wmma_gemm<2><<<dim3(N / 16, K / 16), 32, 0, stream>>>(hacth, w2p, xgl, N, K);
    k_att<<<(N + 255) / 256, 256, 0, stream>>>(xgl, att_src2, att_dst2, as2, ad2, N, 1, K);
    {
        int gb = (ET + 255) / 256;
        k_edge_max <<<gb, 256, 0, stream>>>(src, dst, E, N, 1, as2, ad2, smax2);
        k_edge_sum <<<gb, 256, 0, stream>>>(src, dst, E, N, 1, as2, ad2, smax2, ssum2);
        k_edge_aggr<<<gb, 256, 0, stream>>>(src, dst, E, N, 1, K, as2, ad2, smax2, ssum2, xgl, d_out);
    }
    k_add_bias<<<((size_t)N * K + 255) / 256, 256, 0, stream>>>(d_out, b2, N * K, K);

    // ---- pooling ----
    k_softmax_s<<<(N + 255) / 256, 256, 0, stream>>>(d_out, Ws, bs, sbuf, N);
    k_pack_sT<<<((size_t)K * KPAD + 255) / 256, 256, 0, stream>>>(sbuf, sT, N);
    k_deg<<<(E + 255) / 256, 256, 0, stream>>>(src, deg, E);
    // out = s^T x via WMMA split-K (16 feature tiles x 16 K-splits, 20 wmma each)
    k_wmma_stx<<<dim3(FIN / 16, KPAD / (STX_KSW * 32)), 32, 0, stream>>>(sT, xT, oacc);
    k_outadj<<<(E + OA_EPB - 1) / OA_EPB, 256, 0, stream>>>(sbuf, src, dst, E, oadj);
    k_ss_den<<<(N + SS_NPB - 1) / SS_NPB, 256, 0, stream>>>(sbuf, deg, N, ssb, denb);
    k_finalize<<<1, 256, 0, stream>>>(oadj, ssb, denb, oacc, d_out);

    (void)in_sizes; (void)n_in; (void)out_size; (void)ws_size;
}
